// ShortRangeIntermolecularConsistencyLoss_15607911153869
// MI455X (gfx1250) — compile-verified, matrix operand order checked by reading, not run on hardware
//
#include <hip/hip_runtime.h>
#include <hip/hip_bf16.h>
#include <stdint.h>

#define NATOMS 131072
#define HD 256
#define NF 4096

typedef __attribute__((ext_vector_type(16))) __bf16 v16bf;
typedef __attribute__((ext_vector_type(8)))  float  v8f;

// ---- workspace layout (float offsets) ----
constexpr size_t OFF_SEG_ALL = 0;                       // F*H fragment sums (all atoms) -> overwritten with valid-means
constexpr size_t OFF_SEG_INV = 1048576;                 // F*H fragment sums (invalid atoms only)
constexpr size_t OFF_INTERS  = 2097152;                 // 257 (+3 pad): sum of normalized combined rows
constexpr size_t OFF_SCAL    = 2097412;                 // [0]=w0 [1]=devSum [2]=vecDevSum [3]=interT [4]=validCnt
constexpr size_t OFF_FRAGCNT = 2097420;                 // F
constexpr size_t OFF_SEGMAG  = OFF_FRAGCNT + NF;        // F : sum vec_mag (all)
constexpr size_t OFF_SEGVF   = OFF_SEGMAG + NF;         // F : sum vector_features (valid)
constexpr size_t OFF_MEANNRM = OFF_SEGVF + NF;          // F : ||valid mean||
constexpr size_t OFF_MEANVF  = OFF_MEANNRM + NF;        // F : valid mean of vector_features
constexpr size_t ZERO_END    = OFF_MEANVF + NF;
constexpr size_t OFF_VECMAG  = ZERO_END;                // N
constexpr size_t OFF_FEATSQ  = OFF_VECMAG + NATOMS;     // N : ||feat||^2
constexpr size_t OFF_BPACK   = OFF_FEATSQ + NATOMS;     // 32768 dwords: W1 bf16 in WMMA B layout
constexpr size_t OFF_FEATBF  = OFF_BPACK + 32768;       // N*H bf16 features

__device__ __forceinline__ unsigned short f2bf(float x) {
    unsigned u = __float_as_uint(x);
    u += 0x7fffu + ((u >> 16) & 1u);        // round-to-nearest-even
    return (unsigned short)(u >> 16);
}
__device__ __forceinline__ float bf2f(unsigned s) { return __uint_as_float(s << 16); }
__device__ __forceinline__ float sumsq4(float4 v) { return v.x*v.x + v.y*v.y + v.z*v.z + v.w*v.w; }
__device__ __forceinline__ float wred(float v) {
    v += __shfl_xor(v, 1);  v += __shfl_xor(v, 2);  v += __shfl_xor(v, 4);
    v += __shfl_xor(v, 8);  v += __shfl_xor(v, 16); return v;
}
__device__ __forceinline__ float siluf(float x) { return x / (1.f + __expf(-x)); }

// ---------------- zero accumulators ----------------
__global__ void k_zero(float* W) {
    size_t i = (size_t)blockIdx.x * blockDim.x + threadIdx.x;
    size_t stride = (size_t)gridDim.x * blockDim.x;
    for (; i < ZERO_END; i += stride) W[i] = 0.f;
}

// ---------------- fragment histogram ----------------
__global__ void k_count(const int* __restrict__ fid, float* W) {
    int i = blockIdx.x * 256 + threadIdx.x;
    atomicAdd(&W[OFF_FRAGCNT + fid[i]], 1.f);
}

// ---------------- w0 = sum(Wv[0,:]) ----------------
__global__ void k_w0(const float* __restrict__ Wv, float* W) {
    __shared__ float red[8];
    float v = Wv[threadIdx.x];
    v = wred(v);
    if ((threadIdx.x & 31) == 0) red[threadIdx.x >> 5] = v;
    __syncthreads();
    if (threadIdx.x == 0) { float s = 0.f; for (int i = 0; i < 8; ++i) s += red[i]; W[OFF_SCAL + 0] = s; }
}

// ---------------- pack W1 (f32 row-major) -> bf16 WMMA B-fragments ----------------
// tile (kt in 0..7, nt in 0..15), lane 0..31, dword j 0..7
// lane<16: n = nt*16+lane, K = kt*32 + {0..7, 16..23}; lane>=16: n = nt*16+lane-16, K = kt*32 + {8..15, 24..31}
__global__ void k_packW1(const float* __restrict__ W1, unsigned* __restrict__ bp) {
    int idx  = blockIdx.x * 256 + threadIdx.x;          // 0..32767
    int j    = idx & 7;
    int lane = (idx >> 3) & 31;
    int nt   = (idx >> 8) & 15;
    int kt   = idx >> 12;
    int n    = nt * 16 + (lane & 15);
    int k0   = kt * 32 + (lane >> 4) * 8 + ((j >= 4) ? 16 : 0) + (j & 3) * 2;
    float a = W1[n * HD + k0], b = W1[n * HD + k0 + 1];
    bp[idx] = (unsigned)f2bf(a) | ((unsigned)f2bf(b) << 16);
}

// ---------------- vec_mag: wave per atom ----------------
__global__ void k_vecmag(const float* __restrict__ vs, const int* __restrict__ fid, float* W) {
    __shared__ float vcnt[8];
    int wave = threadIdx.x >> 5, lane = threadIdx.x & 31;
    size_t atom = (size_t)blockIdx.x * 8 + wave;
    const float* p = vs + atom * 768 + lane * 8;
    float s0 = sumsq4(((const float4*)p)[0])         + sumsq4(((const float4*)p)[1]);
    float s1 = sumsq4(((const float4*)(p + 256))[0]) + sumsq4(((const float4*)(p + 256))[1]);
    float s2 = sumsq4(((const float4*)(p + 512))[0]) + sumsq4(((const float4*)(p + 512))[1]);
    s0 = wred(s0); s1 = wred(s1); s2 = wred(s2);
    if (lane == 0) {
        float mag = (sqrtf(s0) + sqrtf(s1) + sqrtf(s2)) * (1.f / 3.f);
        W[OFF_VECMAG + atom] = mag;
        int f = fid[atom];
        atomicAdd(&W[OFF_SEGMAG + f], mag);
        float valid = 0.f;
        if (W[OFF_FRAGCNT + f] >= 3.f) {
            float w0 = W[OFF_SCAL + 0];
            atomicAdd(&W[OFF_SEGVF + f], siluf(mag * w0));
            valid = 1.f;
        }
        vcnt[wave] = valid;
    }
    __syncthreads();
    if (threadIdx.x == 0) {
        float s = 0.f; for (int i = 0; i < 8; ++i) s += vcnt[i];
        if (s > 0.f) atomicAdd(&W[OFF_SCAL + 4], s);
    }
}

// ---------------- main fused LN -> bf16 WMMA GEMM -> silu -> scatter ----------------
__global__ __launch_bounds__(256) void k_main(const float* __restrict__ xs,
                                              const int* __restrict__ fid,
                                              const float* __restrict__ gamma,
                                              const float* __restrict__ beta,
                                              const float* __restrict__ b1,
                                              float* W) {
    __shared__ __align__(16) unsigned char smem[64 * 528 + 64 * 4 * 3]; // xh/feat tile + nsq + fid + valid
    float* nsq   = (float*)(smem + 64 * 528);
    int*   fids  = (int*)(smem + 64 * 528 + 256);
    float* vmask = (float*)(smem + 64 * 528 + 512);

    const int t = threadIdx.x;
    const int blockBase = blockIdx.x * 64;

    // ---- Phase A: LayerNorm, write bf16 A-tile into LDS (row stride 528B, conflict free) ----
    {
        int r = t >> 2, q = t & 3;                        // 4 threads per row
        const float* xrow = xs + (size_t)(blockBase + r) * HD;
        float s = 0.f, s2 = 0.f;
        #pragma unroll
        for (int i = 0; i < 16; ++i) {
            float4 v = *(const float4*)(xrow + q * 4 + i * 16);  // lanes 0..3 contiguous 64B
            s  += v.x + v.y + v.z + v.w;
            s2 += sumsq4(v);
        }
        s  += __shfl_xor(s, 1);  s  += __shfl_xor(s, 2);
        s2 += __shfl_xor(s2, 1); s2 += __shfl_xor(s2, 2);
        float mu  = s * (1.f / HD);
        float inv = rsqrtf(s2 * (1.f / HD) - mu * mu + 1e-5f);
        #pragma unroll
        for (int i = 0; i < 16; ++i) {
            int col = q * 4 + i * 16;
            float4 v  = *(const float4*)(xrow + col);
            float4 g  = *(const float4*)(gamma + col);
            float4 bb = *(const float4*)(beta + col);
            float e0 = (v.x - mu) * inv * g.x + bb.x;
            float e1 = (v.y - mu) * inv * g.y + bb.y;
            float e2 = (v.z - mu) * inv * g.z + bb.z;
            float e3 = (v.w - mu) * inv * g.w + bb.w;
            uint2 pk;
            pk.x = (unsigned)f2bf(e0) | ((unsigned)f2bf(e1) << 16);
            pk.y = (unsigned)f2bf(e2) | ((unsigned)f2bf(e3) << 16);
            *(uint2*)(smem + r * 528 + col * 2) = pk;
        }
        if (t < 64) {
            int f = fid[blockBase + t];
            fids[t]  = f;
            vmask[t] = (W[OFF_FRAGCNT + f] >= 3.f) ? 1.f : 0.f;
            nsq[t]   = 0.f;
        }
    }
    __syncthreads();

    // ---- Phase B: 8 waves x 8 tiles (4 row-tiles x 2 col-tiles each), K=256 in 8 WMMA steps ----
    const int wave = t >> 5, lane = t & 31;
    const int halfW = lane >> 4, ln16 = lane & 15;
    const unsigned* bpack = (const unsigned*)(W + OFF_BPACK);

    v8f acc[8];
    #pragma unroll
    for (int it = 0; it < 8; ++it) {
        const int mt = it >> 1;
        const int nt = wave * 2 + (it & 1);
        v8f c = {0.f, 0.f, 0.f, 0.f, 0.f, 0.f, 0.f, 0.f};
        #pragma unroll
        for (int kt = 0; kt < 8; ++kt) {
            union { uint4 u[2]; v16bf v; } A, B;
            const int row  = mt * 16 + ln16;
            const int koff = kt * 32 + halfW * 8;
            A.u[0] = *(const uint4*)(smem + row * 528 + koff * 2);
            A.u[1] = *(const uint4*)(smem + row * 528 + (koff + 16) * 2);
            const uint4* bp = (const uint4*)(bpack + ((size_t)(kt * 16 + nt) * 32 + lane) * 8);
            B.u[0] = bp[0]; B.u[1] = bp[1];
            c = __builtin_amdgcn_wmma_f32_16x16x32_bf16(false, A.v, false, B.v, (short)0, c, false, false);
        }
        acc[it] = c;
    }
    __syncthreads();   // all A-fragment reads done; reuse tile buffer for bf16 features

    float* segAll = W + OFF_SEG_ALL;
    float* segInv = W + OFF_SEG_INV;
    #pragma unroll
    for (int it = 0; it < 8; ++it) {
        const int mt = it >> 1;
        const int nt = wave * 2 + (it & 1);
        const int n  = nt * 16 + ln16;
        const float bias = b1[n];
        #pragma unroll
        for (int r8 = 0; r8 < 8; ++r8) {
            const int ml = mt * 16 + halfW * 8 + r8;
            float f = siluf(acc[it][r8] + bias);
            *(unsigned short*)(smem + ml * 528 + n * 2) = f2bf(f);
            float sq = f * f;                          // reduce over the 16 columns in this half
            sq += __shfl_xor(sq, 1); sq += __shfl_xor(sq, 2);
            sq += __shfl_xor(sq, 4); sq += __shfl_xor(sq, 8);
            if (ln16 == 0) atomicAdd(&nsq[ml], sq);
            int fr = fids[ml];
            atomicAdd(&segAll[(size_t)fr * HD + n], f);
            if (vmask[ml] == 0.f) atomicAdd(&segInv[(size_t)fr * HD + n], f);
        }
    }
    __syncthreads();

    // ---- coalesced bf16 feature store + per-atom norm^2 ----
    uint4* featDst = (uint4*)(W + OFF_FEATBF) + (size_t)blockBase * 32;
    #pragma unroll
    for (int i = 0; i < 8; ++i) {
        int idx = t + 256 * i;                 // 2048 uint4 per block
        int row = idx >> 5, c4 = idx & 31;
        featDst[idx] = *(const uint4*)(smem + row * 528 + c4 * 16);
    }
    if (t < 64) W[OFF_FEATSQ + blockBase + t] = nsq[t];
}

// ---------------- fragment means + algebraic inter term ----------------
__global__ void k_fragmean(float* W) {
    int wave = threadIdx.x >> 5, lane = threadIdx.x & 31;
    int f = blockIdx.x * 8 + wave;
    float cntAll   = W[OFF_FRAGCNT + f];
    float cntValid = (cntAll >= 3.f) ? cntAll : 0.f;
    float invAll = 1.f / fmaxf(cntAll, 1.f);
    float invVal = 1.f / fmaxf(cntValid, 1.f);
    float*       sa = W + OFF_SEG_ALL + (size_t)f * HD + lane * 8;
    const float* si = W + OFF_SEG_INV + (size_t)f * HD + lane * 8;
    float mean[8], sf[8];
    float nm2 = 0.f, nc2 = 0.f;
    #pragma unroll
    for (int k = 0; k < 8; ++k) {
        float A = sa[k], I = si[k];
        float m = (A - I) * invVal;          // valid-mean
        mean[k] = m; nm2 += m * m;
        float sh = A * invAll;               // all-atom mean (sfrag)
        sf[k] = sh; nc2 += sh * sh;
    }
    #pragma unroll
    for (int k = 0; k < 8; ++k) sa[k] = mean[k];   // overwrite sums with valid-means
    nm2 = wred(nm2); nc2 = wred(nc2);
    float vfrag = W[OFF_SEGMAG + f] * invAll;
    nc2 += vfrag * vfrag;
    float comb = sqrtf(nc2);
    float inv  = 1.f / fmaxf(comb, 1e-12f);
    #pragma unroll
    for (int k = 0; k < 8; ++k) atomicAdd(&W[OFF_INTERS + lane * 8 + k], sf[k] * inv);
    if (lane == 0) {
        atomicAdd(&W[OFF_INTERS + 256], vfrag * inv);
        float cs = comb * inv;                             // ||c_f||
        atomicAdd(&W[OFF_SCAL + 3], cs * cs);              // trace contribution
        W[OFF_MEANNRM + f] = sqrtf(nm2);
        W[OFF_MEANVF + f]  = W[OFF_SEGVF + f] * invVal;
    }
}

// ---------------- pass 2: cosine deviations (scalar + vector branches) ----------------
__global__ void k_pass2(const int* __restrict__ fid, float* W) {
    __shared__ float redD[8], redV[8];
    int wave = threadIdx.x >> 5, lane = threadIdx.x & 31;
    size_t atom = (size_t)blockIdx.x * 8 + wave;
    int f = fid[atom];
    const uint4* fp = (const uint4*)((const unsigned short*)(W + OFF_FEATBF) + atom * HD);
    uint4 fv = fp[lane];
    const float4* mp = (const float4*)(W + OFF_SEG_ALL + (size_t)f * HD + lane * 8);
    float4 m0 = mp[0], m1 = mp[1];
    float d = 0.f; unsigned u;
    u = fv.x; d += bf2f(u & 0xffffu) * m0.x + bf2f(u >> 16) * m0.y;
    u = fv.y; d += bf2f(u & 0xffffu) * m0.z + bf2f(u >> 16) * m0.w;
    u = fv.z; d += bf2f(u & 0xffffu) * m1.x + bf2f(u >> 16) * m1.y;
    u = fv.w; d += bf2f(u & 0xffffu) * m1.z + bf2f(u >> 16) * m1.w;
    d = wred(d);
    if (lane == 0) {
        float valid = (W[OFF_FRAGCNT + f] >= 3.f) ? 1.f : 0.f;
        float fn = sqrtf(W[OFF_FEATSQ + atom]);
        float mn = W[OFF_MEANNRM + f];
        float sim = d / (fmaxf(fn, 1e-12f) * fmaxf(mn, 1e-12f));
        redD[wave] = valid * (1.f - sim);
        float mag = W[OFF_VECMAG + atom];
        float vf  = siluf(mag * W[OFF_SCAL + 0]);
        float mvf = W[OFF_MEANVF + f];
        float a = vf  / fmaxf(fabsf(vf),  1e-12f);
        float b = mvf / fmaxf(fabsf(mvf), 1e-12f);
        redV[wave] = valid * (1.f - a * b);
    }
    __syncthreads();
    if (threadIdx.x == 0) {
        float sd = 0.f, sv = 0.f;
        for (int i = 0; i < 8; ++i) { sd += redD[i]; sv += redV[i]; }
        atomicAdd(&W[OFF_SCAL + 1], sd);
        atomicAdd(&W[OFF_SCAL + 2], sv);
    }
}

// ---------------- finalize ----------------
__global__ void k_final(float* W, float* out) {
    __shared__ float red[8];
    int t = threadIdx.x;
    float v = W[OFF_INTERS + t];
    float s = v * v;
    if (t == 0) { float v2 = W[OFF_INTERS + 256]; s += v2 * v2; }
    s = wred(s);
    if ((t & 31) == 0) red[t >> 5] = s;
    __syncthreads();
    if (t == 0) {
        float tot = 0.f; for (int i = 0; i < 8; ++i) tot += red[i];   // ||sum c_f||^2
        float interT = W[OFF_SCAL + 3];                               // sum ||c_f||^2
        float inter = (tot - interT) / ((float)NF * (float)NF - (float)NF + 1e-6f);
        float vc = fmaxf(W[OFF_SCAL + 4], 1.f);
        float total = W[OFF_SCAL + 1] / vc + W[OFF_SCAL + 2] / vc + 0.2f * inter;
        out[0] = 0.03f * total * 0.05f;
    }
}

extern "C" void kernel_launch(void* const* d_in, const int* in_sizes, int n_in,
                              void* d_out, int out_size, void* d_ws, size_t ws_size,
                              hipStream_t stream) {
    (void)in_sizes; (void)n_in; (void)out_size; (void)ws_size;
    const float* xs    = (const float*)d_in[0];  // scalar_short [N,H]
    const float* vs    = (const float*)d_in[2];  // vector_short [N,3,H]
    const int*   fid   = (const int*)d_in[4];    // fragment_ids [N]
    const float* gamma = (const float*)d_in[5];
    const float* beta  = (const float*)d_in[6];
    const float* W1    = (const float*)d_in[7];
    const float* b1    = (const float*)d_in[8];
    const float* Wv    = (const float*)d_in[9];
    float* W   = (float*)d_ws;
    float* out = (float*)d_out;

    k_zero    <<<2048, 256, 0, stream>>>(W);
    k_count   <<<NATOMS / 256, 256, 0, stream>>>(fid, W);
    k_w0      <<<1, 256, 0, stream>>>(Wv, W);
    k_packW1  <<<128, 256, 0, stream>>>(W1, (unsigned*)(W + OFF_BPACK));
    k_vecmag  <<<NATOMS / 8, 256, 0, stream>>>(vs, fid, W);
    k_main    <<<NATOMS / 64, 256, 0, stream>>>(xs, fid, gamma, beta, b1, W);
    k_fragmean<<<NF / 8, 256, 0, stream>>>(W);
    k_pass2   <<<NATOMS / 8, 256, 0, stream>>>(fid, W);
    k_final   <<<1, 256, 0, stream>>>(W, out);
}
